// TTWindowAttention_81862076661818
// MI455X (gfx1250) — compile-verified, matrix operand order checked by reading, not run on hardware
//
#include <hip/hip_runtime.h>

typedef __attribute__((ext_vector_type(16))) _Float16 v16h;
typedef __attribute__((ext_vector_type(8)))  _Float16 v8h;
typedef __attribute__((ext_vector_type(8)))  float    v8f;

#define WMMA_F32_F16(a, b, c) \
  __builtin_amdgcn_wmma_f32_16x16x32_f16(false, (a), false, (b), (short)0, (c), false, false)

namespace {
constexpr int kN   = 49;     // tokens per window
constexpr int kC   = 384;    // channels
constexpr int kH   = 12;     // heads
constexpr int kHD  = 32;     // head dim
constexpr int kNW  = 1024;   // windows per image
constexpr int kNP  = 64;     // padded rows (4 x 16)
constexpr int kKT  = 12;     // k-tiles of 32 over C=384

constexpr int XS_STR = 392;  // halfs, padded row stride for x/q/k/o (384 + 8)
constexpr int VT_STR = 72;   // halfs, padded row stride for V^T (64 + 8)
constexpr int S_STR  = 65;   // floats, score row stride
constexpr int P_STR  = 72;   // halfs, prob row stride

// dynamic-LDS byte offsets (all 16B aligned)
constexpr int OFF_XS = 0;                               // x (f16), reused as O after attention
constexpr int OFF_QS = OFF_XS + kNP * XS_STR * 2;       // Q (pre-scaled f16)
constexpr int OFF_KS = OFF_QS + kNP * XS_STR * 2;       // K (row major f16)
constexpr int OFF_VT = OFF_KS + kNP * XS_STR * 2;       // V transposed [384][72] f16
constexpr int OFF_S  = OFF_VT + kC  * VT_STR * 2;       // scores f32 [64][65]
constexpr int OFF_P  = OFF_S  + kNP * S_STR  * 4;       // probs  f16 [64][72]
constexpr int SMEM_BYTES = OFF_P + kNP * P_STR * 2;     // = 231,680 B (< 320KB/WGP)

// swizzled-weight workspace: [kt][col][32 halfs in frag order], qkv then proj
constexpr int SW_QKV_ELEMS = kKT * (3 * kC) * 32;       // 442,368 halfs
constexpr int SW_PRJ_ELEMS = kKT * kC * 32;             // 147,456 halfs
constexpr size_t SW_BYTES  = (size_t)(SW_QKV_ELEMS + SW_PRJ_ELEMS) * 2;  // ~1.13 MB
}

// A/B fragment for v_wmma_f32_16x16x32_f16 from a row-major f16 LDS matrix.
// Caller passes pointer already at: row_base + kb + base1 (base1 = 8 for lanes 16..31).
// Elements 0..7 = K{base1..base1+7}, elements 8..15 = K{base1+16..base1+23}: two 16B loads.
__device__ __forceinline__ v16h ld_frag(const _Float16* p) {
  union { v16h v; v8h h[2]; } u;
  u.h[0] = *(const v8h*)(p);
  u.h[1] = *(const v8h*)(p + 16);
  return u.v;
}

// B fragment from the pre-swizzled f16 weight array: 32B contiguous per lane.
__device__ __forceinline__ v16h ld_frag_sw(const _Float16* __restrict__ W2, int ld,
                                           int kt, int col, int lhalf) {
  const _Float16* p = W2 + (((size_t)kt * ld + col) << 5) + (lhalf << 4);
  union { v16h v; v8h h[2]; } u;
  u.h[0] = *(const v8h*)(p);
  u.h[1] = *(const v8h*)(p + 8);
  return u.v;
}

// Fallback: B fragment from row-major f32 global weights, converted in-register.
__device__ __forceinline__ v16h ld_frag_gT(const float* __restrict__ W, int ld, int col, int k0) {
  v16h r;
#pragma unroll
  for (int e = 0; e < 8; ++e) r[e]     = (_Float16)W[(k0 + e) * ld + col];
#pragma unroll
  for (int e = 0; e < 8; ++e) r[8 + e] = (_Float16)W[(k0 + 16 + e) * ld + col];
  return r;
}

// One-time weight convert+swizzle into workspace (f32 -> f16, fragment order).
__global__ __launch_bounds__(256)
void swizzle_weights_kernel(const float* __restrict__ qkv_w,
                            const float* __restrict__ proj_w,
                            _Float16* __restrict__ dst) {
  const int idx = blockIdx.x * 256 + threadIdx.x;
  if (idx >= SW_QKV_ELEMS + SW_PRJ_ELEMS) return;
  const int kbase[4] = {0, 16, 8, 24};  // frag element order within a 32-k tile
  if (idx < SW_QKV_ELEMS) {
    const int e = idx & 31, rest = idx >> 5;
    const int col = rest % (3 * kC), kt = rest / (3 * kC);
    const int k = kt * 32 + kbase[e >> 3] + (e & 7);
    dst[idx] = (_Float16)qkv_w[k * (3 * kC) + col];
  } else {
    const int j = idx - SW_QKV_ELEMS;
    const int e = j & 31, rest = j >> 5;
    const int col = rest % kC, kt = rest / kC;
    const int k = kt * 32 + kbase[e >> 3] + (e & 7);
    dst[idx] = (_Float16)proj_w[k * kC + col];
  }
}

template <bool SW>
__global__ __launch_bounds__(256)
void win_attn_fused_kernel(const float* __restrict__ x,
                           const float* __restrict__ mask,
                           const float* __restrict__ qkv_w,
                           const _Float16* __restrict__ qkv_w_sw,
                           const float* __restrict__ qkv_b,
                           const float* __restrict__ rel_bias,
                           const float* __restrict__ proj_w,
                           const _Float16* __restrict__ proj_w_sw,
                           const float* __restrict__ proj_b,
                           float* __restrict__ out) {
  extern __shared__ char smem[];
  _Float16* xs = (_Float16*)(smem + OFF_XS);  // x, later O
  _Float16* qs = (_Float16*)(smem + OFF_QS);
  _Float16* ks = (_Float16*)(smem + OFF_KS);
  _Float16* vt = (_Float16*)(smem + OFF_VT);
  float*    Ss = (float*)   (smem + OFF_S);
  _Float16* Ps = (_Float16*)(smem + OFF_P);

  const int tid   = threadIdx.x;
  const int wave  = tid >> 5;
  const int lane  = tid & 31;
  const int lhalf = lane >> 4;    // which half-wave
  const int lcol  = lane & 15;    // N-col (B/C/D) or M-row (A) within a tile
  const int base1 = lhalf * 8;    // A/B fragment K sub-base
  const int mrow0 = lhalf * 8;    // C/D fragment row base
  const int blk   = blockIdx.x;
  const int wid   = blk & (kNW - 1);
  const float scale = 0.17677669529663687f;  // 32^-0.5, folded into Q

  // ---------- Stage 0: x -> LDS as f16, rows >= 49 zeroed ----------
  for (int idx = tid; idx < kNP * kC; idx += 256) {
    int r = idx / kC, c = idx - r * kC;
    float v = (r < kN) ? x[((size_t)blk * kN + r) * kC + c] : 0.0f;
    xs[r * XS_STR + c] = (_Float16)v;
  }
  __syncthreads();

  // ---------- Stage 1: QKV GEMM  [64x384] @ [384x1152] ----------
  for (int nt = wave; nt < 72; nt += 8) {
    const int ncol = nt * 16 + lcol;  // 0..1151
    v8f accs[4] = {{}, {}, {}, {}};
#pragma unroll 1
    for (int kt = 0; kt < kKT; ++kt) {
      v16h b;
      if constexpr (SW) {
        b = ld_frag_sw(qkv_w_sw, 3 * kC, kt, ncol, lhalf);
        if (kt + 1 < kKT)
          __builtin_prefetch(qkv_w_sw + (((size_t)(kt + 1) * (3 * kC) + ncol) << 5), 0, 1);
      } else {
        b = ld_frag_gT(qkv_w, 3 * kC, ncol, kt * 32 + base1);
      }
      const int k0 = kt * 32 + base1;
#pragma unroll
      for (int mt = 0; mt < 4; ++mt) {
        v16h a = ld_frag(xs + (mt * 16 + lcol) * XS_STR + k0);
        accs[mt] = WMMA_F32_F16(a, b, accs[mt]);
      }
    }
    const float bias = qkv_b[ncol];
    // region split is wave-uniform in nt (16-col tiles never straddle 384 bounds)
    if (nt < 24) {            // ---- Q (pre-scaled) ----
#pragma unroll
      for (int mt = 0; mt < 4; ++mt)
#pragma unroll
        for (int i = 0; i < 8; ++i) {
          const int m = mt * 16 + mrow0 + i;
          qs[m * XS_STR + ncol] = (_Float16)((accs[mt][i] + bias) * scale);
        }
    } else if (nt < 48) {     // ---- K (row major) ----
      const int c = ncol - kC;
#pragma unroll
      for (int mt = 0; mt < 4; ++mt)
#pragma unroll
        for (int i = 0; i < 8; ++i) {
          const int m = mt * 16 + mrow0 + i;
          ks[m * XS_STR + c] = (_Float16)(accs[mt][i] + bias);
        }
    } else {                  // ---- V (transposed) ----
      const int c = ncol - 2 * kC;
#pragma unroll
      for (int mt = 0; mt < 4; ++mt)
#pragma unroll
        for (int i = 0; i < 8; ++i) {
          const int m = mt * 16 + mrow0 + i;
          vt[c * VT_STR + m] = (_Float16)(accs[mt][i] + bias);
        }
    }
  }
  __syncthreads();

  // ---------- Stage 2: per-head attention ----------
  for (int h = 0; h < kH; ++h) {
    const int hoff = h * kHD;

    // S = Q @ K^T  (16 tiles; 2 per wave; K^T B-frag reads K rows contiguously)
#pragma unroll
    for (int s = 0; s < 2; ++s) {
      const int t  = wave * 2 + s;
      const int mt = t >> 2, ntt = t & 3;
      v16h a = ld_frag(qs + (mt  * 16 + lcol) * XS_STR + hoff + base1);
      v16h b = ld_frag(ks + (ntt * 16 + lcol) * XS_STR + hoff + base1);
      v8f acc = {};
      acc = WMMA_F32_F16(a, b, acc);
      const int n = ntt * 16 + lcol;
#pragma unroll
      for (int i = 0; i < 8; ++i)
        Ss[(mt * 16 + mrow0 + i) * S_STR + n] = acc[i];
    }
    __syncthreads();

    // softmax rows (+ rel_bias + shift mask); padded cols killed, padded rows zeroed
    if (tid < kNP) {
      _Float16* Pr = Ps + tid * P_STR;
      if (tid < kN) {
        float* Sr = Ss + tid * S_STR;
        const float* br = rel_bias + ((size_t)h   * kN + tid) * kN;
        const float* mr = mask     + ((size_t)wid * kN + tid) * kN;
        float mx = -1e30f;
        for (int j = 0; j < kN; ++j) {
          float v = Sr[j] + br[j] + mr[j];
          Sr[j] = v;
          mx = fmaxf(mx, v);
        }
        float sum = 0.0f;
        for (int j = 0; j < kN; ++j) {
          float e = __expf(Sr[j] - mx);
          Sr[j] = e;
          sum += e;
        }
        const float inv = 1.0f / sum;
        for (int j = 0; j < kNP; ++j)
          Pr[j] = (j < kN) ? (_Float16)(Sr[j] * inv) : (_Float16)0.0f;
      } else {
        for (int j = 0; j < kNP; ++j) Pr[j] = (_Float16)0.0f;
      }
    }
    __syncthreads();

    // O = P @ V  (8 tiles; 1 per wave; V^T B-frag contiguous)
    {
      const int mt = wave >> 1, ntt = wave & 1;
      v8f acc = {};
#pragma unroll
      for (int kk = 0; kk < 2; ++kk) {
        const int k0 = kk * 32 + base1;
        v16h a = ld_frag(Ps + (mt * 16 + lcol) * P_STR + k0);
        v16h b = ld_frag(vt + (hoff + ntt * 16 + lcol) * VT_STR + k0);
        acc = WMMA_F32_F16(a, b, acc);
      }
      const int n = hoff + ntt * 16 + lcol;
#pragma unroll
      for (int i = 0; i < 8; ++i)
        xs[(mt * 16 + mrow0 + i) * XS_STR + n] = (_Float16)acc[i];  // O reuses xs
    }
    __syncthreads();
  }

  // ---------- Stage 3: projection  [64x384] @ [384x384] + bias ----------
  for (int nt = wave; nt < 24; nt += 8) {
    const int ncol = nt * 16 + lcol;  // 0..383
    v8f accs[4] = {{}, {}, {}, {}};
#pragma unroll 1
    for (int kt = 0; kt < kKT; ++kt) {
      v16h b;
      if constexpr (SW) {
        b = ld_frag_sw(proj_w_sw, kC, kt, ncol, lhalf);
        if (kt + 1 < kKT)
          __builtin_prefetch(proj_w_sw + (((size_t)(kt + 1) * kC + ncol) << 5), 0, 1);
      } else {
        b = ld_frag_gT(proj_w, kC, ncol, kt * 32 + base1);
      }
      const int k0 = kt * 32 + base1;
#pragma unroll
      for (int mt = 0; mt < 4; ++mt) {
        v16h a = ld_frag(xs + (mt * 16 + lcol) * XS_STR + k0);
        accs[mt] = WMMA_F32_F16(a, b, accs[mt]);
      }
    }
    const float pb = proj_b[ncol];
#pragma unroll
    for (int mt = 0; mt < 4; ++mt) {
#pragma unroll
      for (int i = 0; i < 8; ++i) {
        const int m = mt * 16 + mrow0 + i;
        if (m < kN)
          out[((size_t)blk * kN + m) * kC + ncol] = accs[mt][i] + pb;
      }
    }
  }
}

extern "C" void kernel_launch(void* const* d_in, const int* in_sizes, int n_in,
                              void* d_out, int out_size, void* d_ws, size_t ws_size,
                              hipStream_t stream) {
  (void)in_sizes; (void)n_in; (void)out_size;
  const float* x        = (const float*)d_in[0];
  const float* mask     = (const float*)d_in[1];
  const float* qkv_w    = (const float*)d_in[2];
  const float* qkv_b    = (const float*)d_in[3];
  const float* rel_bias = (const float*)d_in[4];
  const float* proj_w   = (const float*)d_in[5];
  const float* proj_b   = (const float*)d_in[6];
  float* out = (float*)d_out;

  if (ws_size >= SW_BYTES && d_ws != nullptr) {
    _Float16* wsw = (_Float16*)d_ws;
    const int total = SW_QKV_ELEMS + SW_PRJ_ELEMS;
    swizzle_weights_kernel<<<dim3((total + 255) / 256), dim3(256), 0, stream>>>(
        qkv_w, proj_w, wsw);
    win_attn_fused_kernel<true><<<dim3(8192), dim3(256), SMEM_BYTES, stream>>>(
        x, mask, qkv_w, wsw, qkv_b, rel_bias, proj_w, wsw + SW_QKV_ELEMS, proj_b, out);
  } else {
    win_attn_fused_kernel<false><<<dim3(8192), dim3(256), SMEM_BYTES, stream>>>(
        x, mask, qkv_w, nullptr, qkv_b, rel_bias, proj_w, nullptr, proj_b, out);
  }
}
